// Evolution_69776038691012
// MI455X (gfx1250) — matrix operand order repaired and checked by reading, not compiled, not run on hardware
//
#include <hip/hip_runtime.h>
#include <hip/hip_bf16.h>

// ---------------------------------------------------------------------------
// DeepSnake evolution head for MI455X (gfx1250, wave32).
// All GEMM-shaped work (circular convs via on-the-fly im2col, fusion, pred MLP)
// runs on v_wmma_f32_16x16x32_bf16 with f32 accumulation.
// K-loops are software-pipelined (double-buffered A/B fragments); wave tile is
// 16 rows x 64 cols (NT=4) so the pipeline fits in registers without spills.
// ---------------------------------------------------------------------------

typedef __bf16 bf16_t;
typedef __attribute__((ext_vector_type(8)))  __bf16 v8bf;
typedef __attribute__((ext_vector_type(16))) __bf16 v16bf;
typedef __attribute__((ext_vector_type(8)))  float  v8f;

#define NPOLY 128
#define PPTS  128
#define MTOT  (NPOLY * PPTS)   // 16384 rows in every GEMM

__device__ __forceinline__ v16bf bfcat(v8bf lo, v8bf hi) {
  return __builtin_shufflevector(lo, hi, 0,1,2,3,4,5,6,7,8,9,10,11,12,13,14,15);
}

// =====================  weight prep: f32 -> bf16  ==========================

// conv weights: in (nmat, cout, cin, 9) -> out (nmat, cout, 9*cpad), k = tap*cpad + c
__global__ void prep_conv_w(const float* __restrict__ in, bf16_t* __restrict__ out,
                            int nmat, int cout, int cin, int cpad) {
  long idx = (long)blockIdx.x * blockDim.x + threadIdx.x;
  long total = (long)nmat * cout * 9 * cpad;
  if (idx >= total) return;
  int cp  = (int)(idx % cpad);
  long r  = idx / cpad;
  int tap = (int)(r % 9);  r /= 9;
  int co  = (int)(r % cout);
  int mat = (int)(r / cout);
  float v = 0.f;
  if (cp < cin) v = in[(((long)mat * cout + co) * cin + cp) * 9 + tap];
  out[idx] = (bf16_t)v;
}

__global__ void convert_bf16(const float* __restrict__ in, bf16_t* __restrict__ out, long n) {
  long i = (long)blockIdx.x * blockDim.x + threadIdx.x;
  if (i < n) out[i] = (bf16_t)in[i];
}

// =====================  bilinear sampling -> X0 (m, 96)  ===================
// ix = x - 0.5, iy = y - 0.5 (exact simplification of the grid_sample math).
// ch 0..63: bilinear cnn_feature; ch 64..65: (can - min)*RO; ch 66..95: 0.

__global__ void sample_kernel(const float* __restrict__ feat,    // (4,64,128,128)
                              const float* __restrict__ poly,    // (N,P,2)
                              const float* __restrict__ cansrc,  // (N,P,2)
                              const float* __restrict__ minv,    // (N,2) or null
                              const int*   __restrict__ ind,     // (N)
                              bf16_t* __restrict__ X0) {          // (M,96)
  int m = blockIdx.x * blockDim.x + threadIdx.x;   // 0..16383
  int c = blockIdx.y;                               // 0..95
  int n = m >> 7;
  float val = 0.f;
  if (c < 64) {
    float x = poly[2 * m], y = poly[2 * m + 1];
    float ix = x - 0.5f, iy = y - 0.5f;
    float fx0 = floorf(ix), fy0 = floorf(iy);
    int x0 = (int)fx0, y0 = (int)fy0;
    float wx1 = ix - fx0, wy1 = iy - fy0;
    float wx0 = 1.f - wx1, wy0 = 1.f - wy1;
    int b = ind[n];
    const float* fc = feat + ((long)b * 64 + c) * 128 * 128;
    auto gat = [&](int yy, int xx) -> float {
      bool ok = (xx >= 0) & (xx < 128) & (yy >= 0) & (yy < 128);
      int yc = yy < 0 ? 0 : (yy > 127 ? 127 : yy);
      int xc = xx < 0 ? 0 : (xx > 127 ? 127 : xx);
      float t = fc[yc * 128 + xc];
      return ok ? t : 0.f;
    };
    val = gat(y0, x0) * wy0 * wx0 + gat(y0, x0 + 1) * wy0 * wx1 +
          gat(y0 + 1, x0) * wy1 * wx0 + gat(y0 + 1, x0 + 1) * wy1 * wx1;
  } else if (c < 66) {
    int d = c - 64;
    float cp = cansrc[2 * m + d];
    if (minv) cp -= minv[2 * n + d];
    val = cp * 4.0f;   // RO
  }
  X0[(long)m * 96 + c] = (bf16_t)val;
}

// =================  circular-conv GEMM (WMMA bf16)  ========================
// Y(m, 128) = BN(relu(conv(X) + bias)) [+ skip].  K = TAPS*CPAD, im2col on the fly.
// Wave tile: 16 rows x 64 cols (NT=4).  blockIdx.y selects the 64-col group.

template <int CPAD, int TAPS>
__global__ void conv_gemm_kernel(const bf16_t* __restrict__ X,    // (M, CPAD)
                                 const bf16_t* __restrict__ Wt,   // (128, TAPS*CPAD)
                                 const float* __restrict__ bias,
                                 const float* __restrict__ bng, const float* __restrict__ bnb,
                                 const float* __restrict__ bnm, const float* __restrict__ bnv,
                                 const bf16_t* __restrict__ skip, // (M,128) or null
                                 bf16_t* __restrict__ Y,          // (M,128)
                                 int dilation) {
  constexpr int KTOT = TAPS * CPAD;
  constexpr int CB   = CPAD / 32;     // k-blocks per tap
  constexpr int KB   = TAPS * CB;     // total k-blocks
  constexpr int NT   = 4;
  const int lane  = threadIdx.x & 31;
  const int wave  = threadIdx.x >> 5;
  const int mtile = blockIdx.x * (blockDim.x >> 5) + wave;
  const int m0 = mtile * 16;          // rows m0..m0+15, all same poly n (16 | 128)
  const int n  = m0 >> 7;
  const int p0 = m0 & 127;
  const int rA   = lane & 15;         // A row / B col lane index
  const int half = lane >> 4;
  const int colbase = blockIdx.y * (NT * 16);

  auto loadA = [&](int kb) -> v16bf {
    int tap = kb / CB;
    int cb  = (kb % CB) * 32;
    int pp  = (p0 + rA + (tap - TAPS / 2) * dilation) & 127;   // circular P=128
    const bf16_t* arow = X + (long)(n * 128 + pp) * CPAD + cb + half * 8;
    return bfcat(*reinterpret_cast<const v8bf*>(arow),
                 *reinterpret_cast<const v8bf*>(arow + 16));
  };
  auto loadB = [&](int kb, int t) -> v16bf {
    const bf16_t* brow = Wt + (long)(colbase + t * 16 + rA) * KTOT + kb * 32 + half * 8;
    return bfcat(*reinterpret_cast<const v8bf*>(brow),
                 *reinterpret_cast<const v8bf*>(brow + 16));
  };

  v8f acc[NT] = {};
  v16bf aCur = loadA(0);
  v16bf bCur[NT];
#pragma unroll
  for (int t = 0; t < NT; ++t) bCur[t] = loadB(0, t);

  for (int kb = 0; kb < KB - 1; ++kb) {
    v16bf aN = loadA(kb + 1);          // issue next-block loads first
    v16bf bN[NT];
#pragma unroll
    for (int t = 0; t < NT; ++t) bN[t] = loadB(kb + 1, t);
#pragma unroll
    for (int t = 0; t < NT; ++t)       // compute on current block while loads fly
      acc[t] = __builtin_amdgcn_wmma_f32_16x16x32_bf16(false, aCur, false, bCur[t],
                                                       (short)0, acc[t], false, false);
    aCur = aN;
#pragma unroll
    for (int t = 0; t < NT; ++t) bCur[t] = bN[t];
  }
#pragma unroll
  for (int t = 0; t < NT; ++t)
    acc[t] = __builtin_amdgcn_wmma_f32_16x16x32_bf16(false, aCur, false, bCur[t],
                                                     (short)0, acc[t], false, false);

#pragma unroll
  for (int t = 0; t < NT; ++t) {
    int col = colbase + t * 16 + rA;
    float bs = bias[col];
    float sc = bng[col] * rsqrtf(bnv[col] + 1e-5f);
    float sh = bnb[col] - bnm[col] * sc;
#pragma unroll
    for (int r = 0; r < 8; ++r) {
      int row = m0 + r + half * 8;
      float v = fmaxf(acc[t][r] + bs, 0.f) * sc + sh;
      if (skip) v += (float)skip[(long)row * 128 + col];
      Y[(long)row * 128 + col] = (bf16_t)v;
    }
  }
}

// =================  1x1 GEMM (fusion / pred1 / pred2)  =====================
// MODE 0: A = 8-state concat (8,M,128), plain epilogue (fusion)
// MODE 1: A = [g(n,256) || states], relu epilogue (pred1)
// MODE 2: A = dense (M,KTOT), relu epilogue (pred2)
// Wave tile: 16 rows x 64 cols (NT=4); blockIdx.y selects the 64-col group.

template <int KTOT, int MODE>
__global__ void mm_gemm_kernel(const bf16_t* __restrict__ A0,
                               const bf16_t* __restrict__ Gv,
                               const bf16_t* __restrict__ Wt,   // (cout, KTOT)
                               const float* __restrict__ bias,
                               bf16_t* __restrict__ Y, int cout) {
  constexpr int KB = KTOT / 32;
  constexpr int NT = 4;
  const int lane  = threadIdx.x & 31;
  const int wave  = threadIdx.x >> 5;
  const int mtile = blockIdx.x * (blockDim.x >> 5) + wave;
  const int m0 = mtile * 16;
  const int n  = m0 >> 7;
  const int rA   = lane & 15;
  const int half = lane >> 4;
  const int colbase = blockIdx.y * (NT * 16);

  auto loadA = [&](int kb) -> v16bf {
    int k0 = kb * 32;
    const bf16_t* aptr;
    if (MODE == 1 && k0 < 256) {
      aptr = Gv + (long)n * 256 + k0 + half * 8;          // broadcast over rows
    } else if (MODE == 2) {
      aptr = A0 + (long)(m0 + rA) * KTOT + k0 + half * 8;
    } else {
      int kk = (MODE == 1) ? (k0 - 256) : k0;
      int s = kk >> 7, coff = kk & 127;
      aptr = A0 + ((long)s * MTOT + (m0 + rA)) * 128 + coff + half * 8;
    }
    return bfcat(*reinterpret_cast<const v8bf*>(aptr),
                 *reinterpret_cast<const v8bf*>(aptr + 16));
  };
  auto loadB = [&](int kb, int t) -> v16bf {
    const bf16_t* brow = Wt + (long)(colbase + t * 16 + rA) * KTOT + kb * 32 + half * 8;
    return bfcat(*reinterpret_cast<const v8bf*>(brow),
                 *reinterpret_cast<const v8bf*>(brow + 16));
  };

  v8f acc[NT] = {};
  v16bf aCur = loadA(0);
  v16bf bCur[NT];
#pragma unroll
  for (int t = 0; t < NT; ++t) bCur[t] = loadB(0, t);

  for (int kb = 0; kb < KB - 1; ++kb) {
    v16bf aN = loadA(kb + 1);
    v16bf bN[NT];
#pragma unroll
    for (int t = 0; t < NT; ++t) bN[t] = loadB(kb + 1, t);
#pragma unroll
    for (int t = 0; t < NT; ++t)
      acc[t] = __builtin_amdgcn_wmma_f32_16x16x32_bf16(false, aCur, false, bCur[t],
                                                       (short)0, acc[t], false, false);
    aCur = aN;
#pragma unroll
    for (int t = 0; t < NT; ++t) bCur[t] = bN[t];
  }
#pragma unroll
  for (int t = 0; t < NT; ++t)
    acc[t] = __builtin_amdgcn_wmma_f32_16x16x32_bf16(false, aCur, false, bCur[t],
                                                     (short)0, acc[t], false, false);

#pragma unroll
  for (int t = 0; t < NT; ++t) {
    int col = colbase + t * 16 + rA;
    float bs = bias[col];
#pragma unroll
    for (int r = 0; r < 8; ++r) {
      int row = m0 + r + half * 8;
      float v = acc[t][r] + bs;
      if (MODE != 0) v = fmaxf(v, 0.f);
      Y[(long)row * cout + col] = (bf16_t)v;
    }
  }
}

// =====================  reductions & tail  =================================

__global__ void reduce_max_kernel(const bf16_t* __restrict__ F, bf16_t* __restrict__ Gv) {
  int idx = blockIdx.x * blockDim.x + threadIdx.x;   // n*256 + c
  if (idx >= NPOLY * 256) return;
  int n = idx >> 8, c = idx & 255;
  float mx = -3.4e38f;
  const bf16_t* base = F + (long)n * PPTS * 256 + c;
  for (int p = 0; p < PPTS; ++p) mx = fmaxf(mx, (float)base[(long)p * 256]);
  Gv[idx] = (bf16_t)mx;
}

__global__ void poly_min_kernel(const float* __restrict__ P_, float* __restrict__ Mn) {
  int idx = blockIdx.x * blockDim.x + threadIdx.x;   // n*2 + d
  if (idx >= NPOLY * 2) return;
  int n = idx >> 1, d = idx & 1;
  float mn = 3.4e38f;
  for (int p = 0; p < PPTS; ++p) mn = fminf(mn, P_[(long)(n * PPTS + p) * 2 + d]);
  Mn[idx] = mn;
}

// pred3 (64 -> 2) + poly update: out = ipoly*RO + offset; next = out/RO
__global__ void pred3_kernel(const bf16_t* __restrict__ H2,   // (M,64)
                             const float* __restrict__ W3,    // (2,64)
                             const float* __restrict__ b3,    // (2)
                             const float* __restrict__ ipoly, // (M,2)
                             float* __restrict__ out_s,       // (M,2)
                             float* __restrict__ next_poly) { // (M,2) or null
  int m = blockIdx.x * blockDim.x + threadIdx.x;
  if (m >= MTOT) return;
  float o0 = b3[0], o1 = b3[1];
  const bf16_t* h = H2 + (long)m * 64;
#pragma unroll 4
  for (int c = 0; c < 64; ++c) {
    float hv = (float)h[c];
    o0 += W3[c] * hv;
    o1 += W3[64 + c] * hv;
  }
  float px = ipoly[2 * m] * 4.0f + o0;
  float py = ipoly[2 * m + 1] * 4.0f + o1;
  out_s[2 * m] = px;
  out_s[2 * m + 1] = py;
  if (next_poly) { next_poly[2 * m] = px * 0.25f; next_poly[2 * m + 1] = py * 0.25f; }
}

// ===========================  driver  ======================================

extern "C" void kernel_launch(void* const* d_in, const int* in_sizes, int n_in,
                              void* d_out, int out_size, void* d_ws, size_t ws_size,
                              hipStream_t stream) {
  (void)in_sizes; (void)n_in; (void)out_size; (void)ws_size;

  const float* feat    = (const float*)d_in[0];
  const float* imgPoly = (const float*)d_in[1];
  const float* canPoly = (const float*)d_in[2];
  const int*   ind     = (const int*)d_in[3];
  const float* headW   = (const float*)d_in[4];
  const float* headB   = (const float*)d_in[5];
  const float* headG   = (const float*)d_in[6];
  const float* headBe  = (const float*)d_in[7];
  const float* headM   = (const float*)d_in[8];
  const float* headV   = (const float*)d_in[9];
  const float* resW    = (const float*)d_in[10];
  const float* resB    = (const float*)d_in[11];
  const float* resG    = (const float*)d_in[12];
  const float* resBe   = (const float*)d_in[13];
  const float* resM    = (const float*)d_in[14];
  const float* resV    = (const float*)d_in[15];
  const float* fusW    = (const float*)d_in[16];
  const float* fusB    = (const float*)d_in[17];
  const float* p1W     = (const float*)d_in[18];
  const float* p1B     = (const float*)d_in[19];
  const float* p2W     = (const float*)d_in[20];
  const float* p2B     = (const float*)d_in[21];
  const float* p3W     = (const float*)d_in[22];
  const float* p3B     = (const float*)d_in[23];
  float* out = (float*)d_out;

  // ---- workspace carve-up (256B aligned slices) ----
  size_t off = 0;
  auto carve = [&](size_t bytes) -> char* {
    char* p = (char*)d_ws + off;
    off += (bytes + 255) & ~(size_t)255;
    return p;
  };
  bf16_t* WbHead  = (bf16_t*)carve(3ull * 128 * 864 * 2);
  bf16_t* WbRes   = (bf16_t*)carve(21ull * 128 * 1152 * 2);
  bf16_t* WbFus   = (bf16_t*)carve(3ull * 256 * 1024 * 2);
  bf16_t* WbP1    = (bf16_t*)carve(3ull * 256 * 1280 * 2);
  bf16_t* WbP2    = (bf16_t*)carve(3ull * 64 * 256 * 2);
  bf16_t* X0      = (bf16_t*)carve((size_t)MTOT * 96 * 2);
  bf16_t* S8      = (bf16_t*)carve(8ull * MTOT * 128 * 2);
  bf16_t* FusedH1 = (bf16_t*)carve((size_t)MTOT * 256 * 2);  // fusion out, then pred1 out
  bf16_t* Gbuf    = (bf16_t*)carve((size_t)NPOLY * 256 * 2);
  bf16_t* H2      = (bf16_t*)carve((size_t)MTOT * 64 * 2);
  float*  CurPoly = (float*)carve((size_t)MTOT * 2 * 4);
  float*  MinV    = (float*)carve((size_t)NPOLY * 2 * 4);

  // ---- weight prep (bf16, WMMA-friendly (cout, K) layout) ----
  {
    long t;
    t = 3L * 128 * 9 * 96;
    prep_conv_w<<<dim3((t + 255) / 256), 256, 0, stream>>>(headW, WbHead, 3, 128, 66, 96);
    t = 21L * 128 * 9 * 128;
    prep_conv_w<<<dim3((t + 255) / 256), 256, 0, stream>>>(resW, WbRes, 21, 128, 128, 128);
    t = 3L * 256 * 1024;
    convert_bf16<<<dim3((t + 255) / 256), 256, 0, stream>>>(fusW, WbFus, t);
    t = 3L * 256 * 1280;
    convert_bf16<<<dim3((t + 255) / 256), 256, 0, stream>>>(p1W, WbP1, t);
    t = 3L * 64 * 256;
    convert_bf16<<<dim3((t + 255) / 256), 256, 0, stream>>>(p2W, WbP2, t);
  }

  const int DIL[7] = {1, 1, 1, 2, 2, 4, 4};
  const dim3 convGrid(128, 2);   // 128 blocks * 8 waves * 2 col-groups = 2048 waves

  for (int s = 0; s < 3; ++s) {
    const float* ip = (s == 0) ? imgPoly : CurPoly;
    const float* cp = (s == 0) ? canPoly : CurPoly;
    const float* mv = (s == 0) ? nullptr : MinV;

    sample_kernel<<<dim3(MTOT / 256, 96), 256, 0, stream>>>(feat, ip, cp, mv, ind, X0);

    // head block -> state 0
    conv_gemm_kernel<96, 9><<<convGrid, 256, 0, stream>>>(
        X0, WbHead + (size_t)s * 128 * 864,
        headB + s * 128, headG + s * 128, headBe + s * 128, headM + s * 128, headV + s * 128,
        nullptr, S8, 1);

    // 7 residual blocks
    for (int i = 0; i < 7; ++i) {
      size_t li = (size_t)s * 7 + i;
      bf16_t* xin = S8 + (size_t)i * MTOT * 128;
      conv_gemm_kernel<128, 9><<<convGrid, 256, 0, stream>>>(
          xin, WbRes + li * 128 * 1152,
          resB + li * 128, resG + li * 128, resBe + li * 128, resM + li * 128, resV + li * 128,
          xin /*skip*/, S8 + (size_t)(i + 1) * MTOT * 128, DIL[i]);
    }

    // fusion 1024 -> 256 (plain), then global max over P
    mm_gemm_kernel<1024, 0><<<dim3(128, 4), 256, 0, stream>>>(
        S8, nullptr, WbFus + (size_t)s * 256 * 1024, fusB + s * 256, FusedH1, 256);
    reduce_max_kernel<<<dim3(NPOLY * 256 / 256), 256, 0, stream>>>(FusedH1, Gbuf);

    // pred1: [g || states] 1280 -> 256, relu (overwrites FusedH1; fused no longer read)
    mm_gemm_kernel<1280, 1><<<dim3(128, 4), 256, 0, stream>>>(
        S8, Gbuf, WbP1 + (size_t)s * 256 * 1280, p1B + s * 256, FusedH1, 256);

    // pred2: 256 -> 64, relu
    mm_gemm_kernel<256, 2><<<dim3(128, 1), 256, 0, stream>>>(
        FusedH1, nullptr, WbP2 + (size_t)s * 64 * 256, p2B + s * 64, H2, 64);

    // pred3 + poly update
    pred3_kernel<<<dim3(MTOT / 256), 256, 0, stream>>>(
        H2, p3W + (size_t)s * 128, p3B + s * 2, ip,
        out + (size_t)s * MTOT * 2, (s < 2) ? CurPoly : nullptr);

    if (s < 2)
      poly_min_kernel<<<dim3(1), 256, 0, stream>>>(CurPoly, MinV);
  }
}
